// Sparse3DRegression_15650860827383
// MI455X (gfx1250) — matrix operand order repaired and checked by reading, not compiled, not run on hardware
//
#include <hip/hip_runtime.h>
#include <stdint.h>

typedef __attribute__((ext_vector_type(16))) _Float16 v16h;
typedef __attribute__((ext_vector_type(8)))  float    v8f;

#define EMPTY_SLOT 0x7FFFFFFF
#define K_AN 0.1f

// ---------------------------------------------------------------- helpers
__device__ __forceinline__ int lower_bound_i32(const int* a, int n, int key) {
  int lo = 0, hi = n;
  while (lo < hi) { int mid = (lo + hi) >> 1; if (a[mid] < key) lo = mid + 1; else hi = mid; }
  return lo;
}
__device__ __forceinline__ unsigned enc_f32(float f) {
  unsigned u = __float_as_uint(f);
  return (u & 0x80000000u) ? ~u : (u | 0x80000000u);
}
__device__ __forceinline__ float dec_f32(unsigned u) {
  unsigned b = (u & 0x80000000u) ? (u & 0x7FFFFFFFu) : ~u;
  return __uint_as_float(b);
}
__device__ __forceinline__ int hash_insert(int* tab, int H, int id) {
  unsigned h = ((unsigned)id * 2654435761u) & (unsigned)(H - 1);
  for (int t = 0; t < H; ++t) {
    int prev = atomicCAS(&tab[h], EMPTY_SLOT, id);
    if (prev == EMPTY_SLOT || prev == id) return (int)h;
    h = (h + 1) & (unsigned)(H - 1);
  }
  return -1;
}

// ---------------------------------------------------------------- fills
__global__ void k_fill_u32(unsigned* p, unsigned v, int n) {
  int i = blockIdx.x * blockDim.x + threadIdx.x; if (i < n) p[i] = v;
}
__global__ void k_fill_u64(unsigned long long* p, unsigned long long v, int n) {
  int i = blockIdx.x * blockDim.x + threadIdx.x; if (i < n) p[i] = v;
}

// ---------------------------------------------------------------- min/max of point coords
__global__ void k_minmax(const float* __restrict__ pts, unsigned* mm, int n) {
  int i = blockIdx.x * blockDim.x + threadIdx.x; if (i >= n) return;
  #pragma unroll
  for (int c = 0; c < 3; ++c) {
    unsigned u = enc_f32(pts[i * 3 + c]);
    atomicMin(&mm[c], u);
    atomicMax(&mm[3 + c], u);
  }
}

// ---------------------------------------------------------------- voxel ids + hash insert + counts
__global__ void k_point_ids(const float* __restrict__ pts, const int* __restrict__ batch,
                            const unsigned* __restrict__ mm, int n, int G,
                            int* table, int* hcnt, int H) {
  int i = blockIdx.x * blockDim.x + threadIdx.x; if (i >= n) return;
  int g[3];
  #pragma unroll
  for (int c = 0; c < 3; ++c) {
    float mn = dec_f32(mm[c]), mx = dec_f32(mm[3 + c]);
    float gf = (pts[i * 3 + c] - mn) / (mx - mn + 1e-6f) * (float)(G - 1);
    int gi = (int)gf;
    g[c] = min(max(gi, 0), G - 1);
  }
  int b = batch[i];
  int id = ((b * G + g[2]) * G + g[1]) * G + g[0];
  int slot = hash_insert(table, H, id);
  if (slot >= 0) atomicAdd(&hcnt[slot], 1);
}

// ---------------------------------------------------------------- downsample candidate ids
__global__ void k_down_cand(const int* __restrict__ ids, int n, int Gin, int Gout, int INVin,
                            int* table, int H) {
  int i = blockIdx.x * blockDim.x + threadIdx.x; if (i >= n) return;
  int id = ids[i];
  if (id >= INVin) return;
  int x = id % Gin; int t = id / Gin;
  int y = t % Gin;  t /= Gin;
  int z = t % Gin;  int b = t / Gin;
  for (int k = 0; k < 27; ++k) {
    int kz = k / 9, ky = (k / 3) % 3, kx = k % 3;
    int oz = z + 1 - kz, oy = y + 1 - ky, ox = x + 1 - kx;
    if ((oz & 1) || (oy & 1) || (ox & 1)) continue;
    if (oz < 0 || oy < 0 || ox < 0) continue;
    int oz2 = oz >> 1, oy2 = oy >> 1, ox2 = ox >> 1;
    if (oz2 >= Gout || oy2 >= Gout || ox2 >= Gout) continue;
    int oid = ((b * Gout + oz2) * Gout + oy2) * Gout + ox2;
    hash_insert(table, H, oid);
  }
}

// ---------------------------------------------------------------- hash -> key list
__global__ void k_collect(const int* __restrict__ table, int H,
                          unsigned long long* keys, int* counter, int cap) {
  int i = blockIdx.x * blockDim.x + threadIdx.x; if (i >= H) return;
  int id = table[i];
  if (id != EMPTY_SLOT) {
    int idx = atomicAdd(counter, 1);
    if (idx < cap) keys[idx] = (((unsigned long long)(unsigned)id) << 32) | (unsigned long long)(unsigned)i;
  }
}

// ---------------------------------------------------------------- bitonic sort step
__global__ void k_bitonic(unsigned long long* keys, int n, int j, int k) {
  int i = blockIdx.x * blockDim.x + threadIdx.x; if (i >= n) return;
  int ixj = i ^ j;
  if (ixj > i) {
    unsigned long long a = keys[i], b = keys[ixj];
    bool up = ((i & k) == 0);
    if ((a > b) == up) { keys[i] = b; keys[ixj] = a; }
  }
}

// ---------------------------------------------------------------- sorted keys -> ids (+counts)
__global__ void k_extract(const unsigned long long* __restrict__ keys, int cap, int INV,
                          int* ids, float* countsF, const int* __restrict__ hcnt) {
  int i = blockIdx.x * blockDim.x + threadIdx.x; if (i >= cap) return;
  unsigned long long key = keys[i];
  if (key == ~0ull) {
    ids[i] = INV;
    if (countsF) countsF[i] = 0.f;
  } else {
    ids[i] = (int)(key >> 32);
    if (countsF) countsF[i] = (float)hcnt[(unsigned)(key & 0xFFFFFFFFull)];
  }
}

// ---------------------------------------------------------------- first conv (ci=1, co=16), scalar
__global__ void k_conv_c1(const int* __restrict__ ids, int n, int G, int INV,
                          const float* __restrict__ cnt, const float* __restrict__ w1,
                          const float* __restrict__ b1, float* __restrict__ out) {
  int i = blockIdx.x * blockDim.x + threadIdx.x; if (i >= n) return;
  int id = ids[i];
  bool m = id < INV;
  int cid = m ? id : 0;
  int x = cid % G; int t = cid / G;
  int y = t % G;   t /= G;
  int z = t % G;   int b = t / G;
  float acc[16];
  #pragma unroll
  for (int c = 0; c < 16; ++c) acc[c] = 0.f;
  for (int k = 0; k < 27; ++k) {
    int dz = k / 9 - 1, dy = (k / 3) % 3 - 1, dx = k % 3 - 1;
    int nz = z + dz, ny = y + dy, nx = x + dx;
    bool inb = m && nz >= 0 && nz < G && ny >= 0 && ny < G && nx >= 0 && nx < G;
    float cf = 0.f;
    if (inb) {
      int nid = ((b * G + nz) * G + ny) * G + nx;
      int pos = lower_bound_i32(ids, n, nid);
      if (pos < n && ids[pos] == nid) cf = cnt[pos];
    }
    if (cf != 0.f) {
      #pragma unroll
      for (int c = 0; c < 16; ++c) acc[c] += cf * w1[k * 16 + c];
    }
  }
  float mf = m ? 1.f : 0.f;
  #pragma unroll
  for (int c = 0; c < 16; ++c) out[(size_t)i * 16 + c] = (acc[c] + b1[c]) * mf;
}

// ---------------------------------------------------------------- AdaNorm + ReLU (wave per row)
__global__ void k_adanorm_relu(float* __restrict__ x, const float* __restrict__ g, int rows, int C) {
  int wave = (blockIdx.x * blockDim.x + threadIdx.x) >> 5;
  int lane = threadIdx.x & 31;
  if (wave >= rows) return;
  float* row = x + (size_t)wave * C;
  float s = 0.f;
  for (int c = lane; c < C; c += 32) s += row[c];
  #pragma unroll
  for (int o = 16; o; o >>= 1) s += __shfl_xor(s, o);
  float mean = s / (float)C;
  float v = 0.f;
  for (int c = lane; c < C; c += 32) { float d = row[c] - mean; v += d * d; }
  #pragma unroll
  for (int o = 16; o; o >>= 1) v += __shfl_xor(v, o);
  float r = rsqrtf(v / (float)C + 1e-5f);
  for (int c = lane; c < C; c += 32) {
    float xn = (row[c] - mean) * r;
    float yv = g[c] * (1.f - K_AN * xn) * xn;
    row[c] = fmaxf(yv, 0.f);
  }
}

// ---------------------------------------------------------------- weight swizzle -> B-fragment layout
// B (K x N) 16-bit layout: lanes 0-15 hold K=kc*32+[0..7,16..23], lanes 16-31 hold K=kc*32+[8..15,24..31]; N = nt*16 + (lane&15)
__global__ void k_wswz(const float* __restrict__ W, int ci, int co, int KC, int NT,
                       _Float16* __restrict__ Wb) {
  int t = blockIdx.x * blockDim.x + threadIdx.x;
  int total = 27 * KC * NT * 32;
  if (t >= total) return;
  int lane = t & 31;
  int nt = (t >> 5) % NT;
  int kc = ((t >> 5) / NT) % KC;
  int k  = (t >> 5) / (NT * KC);
  int N = nt * 16 + (lane & 15);
  int gb = (lane & 16) ? 16 : 0;
  _Float16* dst = Wb + ((size_t)(((k * KC + kc) * NT + nt) * 32 + lane)) * 16;
  #pragma unroll
  for (int h = 0; h < 16; ++h) {
    int kk = kc * 32 + (gb >> 1) + (h < 8 ? h : h + 8);
    float val = (kk < ci && N < co) ? W[((size_t)k * ci + kk) * co + N] : 0.f;
    dst[h] = (_Float16)val;
  }
}

// ---------------------------------------------------------------- gather-GEMM core (WMMA)
// Phase 1: all 27x16 neighbor binary searches in parallel -> LDS table.
// Phase 2: software-pipelined taps; async global->LDS (ASYNCcnt) double-buffered
//          against WMMA consumption. Invalid gathers read a zero row, so the A
//          build is a pure ds_load -> cvt chain (no masking VALU).
// mode 0: submanifold (Gout==Gin), mode 1: strided down (input coords = 2*out + k - 1)
template <int CI, int CO>
__global__ __launch_bounds__(256) void k_gather_gemm(
    const int* __restrict__ out_ids, int nOut, int Gout,
    const int* __restrict__ in_ids, int nIn, int Gin,
    const float* __restrict__ in_feats, const float* __restrict__ zrow,
    const _Float16* __restrict__ Wb, const float* __restrict__ bias,
    int mode, int INVout, float* __restrict__ out_feats) {
  constexpr int KC = (CI + 31) / 32;  // K chunks of 32
  constexpr int NT = CO / 16;         // N tiles of 16
  constexpr int NI = CI / 8;          // async b128 instructions per lane per tap
  constexpr int CH = CI / 2;          // channels staged per lane

  __shared__ float stage[8][2][16][64];   // per-wave double-buffered 16 x 64ch staging
  __shared__ int   srcTab[8][27][16];     // per-wave gather indices

  const int lane = threadIdx.x & 31;
  const int w = threadIdx.x >> 5;
  const int rowBase = (blockIdx.x * 8 + w) * 16;
  const int sr = lane >> 1;   // staging: this lane serves row sr
  const int half = lane & 1;  // and channel half

  __builtin_prefetch(in_ids + (nIn >> 1), 0, 0);  // global_prefetch_b8: warm search root

  // zero-pad K columns [CI..32) once when CI < 32 (async loads never touch them)
  if (CI < 32) {
    const float4 z4 = make_float4(0.f, 0.f, 0.f, 0.f);
    #pragma unroll
    for (int bf = 0; bf < 2; ++bf)
      #pragma unroll
      for (int i = 0; i < (32 - CI) / 8; ++i)
        *(float4*)&stage[w][bf][sr][CI + half * ((32 - CI) / 2) + i * 4] = z4;
  }

  // ---- phase 1: 27 taps x 16 rows of binary searches, spread over 32 lanes ----
  for (int idx = lane; idx < 27 * 16; idx += 32) {
    const int r = idx & 15;
    const int k = idx >> 4;
    int oid = out_ids[rowBase + r];
    bool om = oid < INVout;
    int cid = om ? oid : 0;
    int x = cid % Gout; int t = cid / Gout;
    int y = t % Gout;   t /= Gout;
    int z = t % Gout;   int b = t / Gout;
    int nz, ny, nx;
    if (mode == 0) {
      nz = z + k / 9 - 1; ny = y + (k / 3) % 3 - 1; nx = x + k % 3 - 1;
    } else {
      nz = 2 * z + k / 9 - 1; ny = 2 * y + (k / 3) % 3 - 1; nx = 2 * x + k % 3 - 1;
    }
    int src = -1;
    if (om && nz >= 0 && nz < Gin && ny >= 0 && ny < Gin && nx >= 0 && nx < Gin) {
      int nid = ((b * Gin + nz) * Gin + ny) * Gin + nx;
      int pos = lower_bound_i32(in_ids, nIn, nid);
      if (pos < nIn && in_ids[pos] == nid) src = pos;
    }
    srcTab[w][k][r] = src;  // wave-local LDS: in-order with later ds reads
  }

  // row-valid bits for the final store mask
  const unsigned mb = (unsigned)__ballot(out_ids[rowBase + (lane & 15)] < INVout) & 0xFFFFu;

  v8f acc[NT];
  #pragma unroll
  for (int nt = 0; nt < NT; ++nt)
    #pragma unroll
    for (int j = 0; j < 8; ++j) acc[nt][j] = 0.f;

  // issue one tap's gather: NI async b128 per lane; invalid rows stream the zero row
  auto issue_tap = [&](int k, int buf) {
    int srcR = srcTab[w][k][sr];
    const float* gsrc = (srcR < 0 ? zrow : in_feats + (size_t)srcR * CI) + half * CH;
    unsigned lb = (unsigned)(uintptr_t)&stage[w][buf][sr][half * CH];
    #pragma unroll
    for (int i = 0; i < NI; ++i) {
      unsigned lo = lb + (unsigned)(i * 16);
      unsigned long long ga = (unsigned long long)(uintptr_t)(gsrc + i * 4);
      asm volatile("global_load_async_to_lds_b128 %0, %1, off"
                   :: "v"(lo), "v"(ga) : "memory");
    }
  };

  // ---- phase 2: software-pipelined taps ----
  issue_tap(0, 0);
  for (int k = 0; k < 27; ++k) {
    const int buf = k & 1;
    if (k + 1 < 27) {
      issue_tap(k + 1, buf ^ 1);
      // wait for tap k's NI loads (async-to-LDS completes in order); tap k+1 stays in flight
      asm volatile("s_wait_asynccnt %0" :: "i"(NI) : "memory");
    } else {
      asm volatile("s_wait_asynccnt 0" ::: "memory");
    }

    #pragma unroll
    for (int kc = 0; kc < KC; ++kc) {
      // B fragments first: overlap their latency with the A-fragment cvt chain
      v16h bfr[NT];
      #pragma unroll
      for (int nt = 0; nt < NT; ++nt)
        bfr[nt] = *((const v16h*)Wb + ((size_t)(k * KC + kc) * NT + nt) * 32 + lane);

      // A fragment: 16x32 f16 layout (lanes 0-15: K 0..7 & 16..23; lanes 16-31: 8..15 & 24..31)
      const float* rp = &stage[w][buf][lane & 15][kc * 32];
      const int group = (lane & 16) ? 8 : 0;
      v16h a;
      #pragma unroll
      for (int h = 0; h < 8; ++h)  a[h] = (_Float16)rp[group + h];
      #pragma unroll
      for (int h = 8; h < 16; ++h) a[h] = (_Float16)rp[8 + group + h];

      #pragma unroll
      for (int nt = 0; nt < NT; ++nt)
        acc[nt] = __builtin_amdgcn_wmma_f32_16x16x32_f16(
            false, a, false, bfr[nt], (short)0, acc[nt], false, false);
    }
  }

  // ---- bias, mask, store (D layout: vgpr r -> M=r+(lane>=16?8:0), N=lane&15) ----
  const int moff = (lane & 16) ? 8 : 0;
  #pragma unroll
  for (int nt = 0; nt < NT; ++nt) {
    const float bv = bias[nt * 16 + (lane & 15)];
    #pragma unroll
    for (int r = 0; r < 8; ++r) {
      int M = r + moff;
      float mf = ((mb >> M) & 1u) ? 1.f : 0.f;
      float vv = (acc[nt][r] + bv) * mf;
      out_feats[(size_t)(rowBase + M) * CO + nt * 16 + (lane & 15)] = vv;
    }
  }
}

// ---------------------------------------------------------------- batch max-pool (values >= 0)
__global__ void k_pool(const int* __restrict__ ids, int n, int INV, int C,
                       const float* __restrict__ h, unsigned* __restrict__ pooled, int G) {
  int i = blockIdx.x * blockDim.x + threadIdx.x; if (i >= n) return;
  int id = ids[i];
  if (id >= INV) return;
  int b = id / (G * G * G);
  for (int c = 0; c < C; ++c) {
    float v = h[(size_t)i * C + c];
    atomicMax(&pooled[b * C + c], __float_as_uint(v));  // monotone for v >= 0
  }
}

// ---------------------------------------------------------------- MLP head (4 rows)
__global__ void k_head(const float* __restrict__ pooled,
                       const float* __restrict__ wh, const float* __restrict__ bh,
                       const float* __restrict__ gh,
                       const float* __restrict__ wf1, const float* __restrict__ bf1,
                       const float* __restrict__ wf2, const float* __restrict__ bf2,
                       float* __restrict__ out) {
  __shared__ float sp[4][64];
  __shared__ float s1[4][128];
  __shared__ float red[4][2];
  int t = threadIdx.x;  // 128 threads
  if (t < 64) { for (int r = 0; r < 4; ++r) sp[r][t] = pooled[r * 64 + t]; }
  __syncthreads();
  for (int r = 0; r < 4; ++r) {
    float a = bh[t];
    for (int i = 0; i < 64; ++i) a += sp[r][i] * wh[i * 128 + t];
    s1[r][t] = a;
  }
  __syncthreads();
  if (t < 4) {
    float m = 0.f; for (int i = 0; i < 128; ++i) m += s1[t][i];
    m /= 128.f;
    float v = 0.f; for (int i = 0; i < 128; ++i) { float d = s1[t][i] - m; v += d * d; }
    red[t][0] = m; red[t][1] = rsqrtf(v / 128.f + 1e-5f);
  }
  __syncthreads();
  for (int r = 0; r < 4; ++r) {
    float xn = (s1[r][t] - red[r][0]) * red[r][1];
    float yv = gh[t] * (1.f - K_AN * xn) * xn;
    s1[r][t] = fmaxf(yv, 0.f);
  }
  __syncthreads();
  float h2v[4];
  for (int r = 0; r < 4; ++r) {
    float a = bf1[t];
    for (int i = 0; i < 128; ++i) a += s1[r][i] * wf1[i * 128 + t];
    h2v[r] = fmaxf(a, 0.f);
  }
  __syncthreads();
  for (int r = 0; r < 4; ++r) s1[r][t] = h2v[r];
  __syncthreads();
  if (t < 32) {
    int r = t >> 3, o = t & 7;
    float a = bf2[o];
    for (int i = 0; i < 128; ++i) a += s1[r][i] * wf2[i * 8 + o];
    out[r * 8 + o] = a;
  }
}

// ---------------------------------------------------------------- host orchestration
static void bitonic_sort(unsigned long long* keys, int n, hipStream_t s) {
  int blocks = n / 256;
  for (int k = 2; k <= n; k <<= 1)
    for (int j = k >> 1; j > 0; j >>= 1)
      k_bitonic<<<blocks, 256, 0, s>>>(keys, n, j, k);
}

extern "C" void kernel_launch(void* const* d_in, const int* in_sizes, int n_in,
                              void* d_out, int out_size, void* d_ws, size_t ws_size,
                              hipStream_t stream) {
  (void)in_sizes; (void)n_in; (void)out_size; (void)ws_size;
  const int N = 32768;
  const int G0 = 256, G1 = 128, G2 = 64;
  const int CAP1 = 32768, CAP2 = 65536;
  const int INV0 = 4 * G0 * G0 * G0;   // 67108864
  const int INV1 = 4 * G1 * G1 * G1;   // 8388608
  const int INV2 = 4 * G2 * G2 * G2;   // 1048576

  const float* points = (const float*)d_in[0];
  const int*   batch  = (const int*)d_in[1];
  const float* w1  = (const float*)d_in[2];
  const float* b1  = (const float*)d_in[3];
  const float* g1  = (const float*)d_in[4];
  const float* wd1 = (const float*)d_in[5];
  const float* bd1 = (const float*)d_in[6];
  const float* g2  = (const float*)d_in[7];
  const float* w2  = (const float*)d_in[8];
  const float* b2  = (const float*)d_in[9];
  const float* g3  = (const float*)d_in[10];
  const float* wd2 = (const float*)d_in[11];
  const float* bd2 = (const float*)d_in[12];
  const float* g4  = (const float*)d_in[13];
  const float* w3  = (const float*)d_in[14];
  const float* b3  = (const float*)d_in[15];
  const float* g5  = (const float*)d_in[16];
  const float* wh  = (const float*)d_in[17];
  const float* bh  = (const float*)d_in[18];
  const float* gh  = (const float*)d_in[19];
  const float* wf1 = (const float*)d_in[20];
  const float* bf1 = (const float*)d_in[21];
  const float* wf2 = (const float*)d_in[22];
  const float* bf2 = (const float*)d_in[23];

  char* wsb = (char*)d_ws;
  size_t off = 0;
  auto alloc = [&](size_t bytes) -> void* {
    off = (off + 255) & ~(size_t)255;
    void* p = wsb + off; off += bytes; return p;
  };
  unsigned* mm       = (unsigned*)alloc(6 * 4);
  int* table         = (int*)alloc((size_t)131072 * 4);
  int* hcnt          = (int*)alloc((size_t)131072 * 4);
  int* counter       = (int*)alloc(4);
  unsigned long long* keys = (unsigned long long*)alloc((size_t)65536 * 8);
  float* zrow = (float*)alloc(64 * 4);
  int*   ids0 = (int*)alloc((size_t)N * 4);
  float* cnt0 = (float*)alloc((size_t)N * 4);
  float* h0   = (float*)alloc((size_t)N * 16 * 4);
  int*   ids1 = (int*)alloc((size_t)CAP1 * 4);
  float* h1   = (float*)alloc((size_t)CAP1 * 32 * 4);
  float* h2   = (float*)alloc((size_t)CAP1 * 32 * 4);
  int*   ids2 = (int*)alloc((size_t)CAP2 * 4);
  float* h3   = (float*)alloc((size_t)CAP2 * 64 * 4);
  float* h4   = (float*)alloc((size_t)CAP2 * 64 * 4);
  unsigned* pooled = (unsigned*)alloc(4 * 64 * 4);
  _Float16* Wb1 = (_Float16*)alloc((size_t)27648 * 2);
  _Float16* Wb2 = (_Float16*)alloc((size_t)27648 * 2);
  _Float16* Wb3 = (_Float16*)alloc((size_t)55296 * 2);
  _Float16* Wb4 = (_Float16*)alloc((size_t)221184 * 2);

  // ---- voxelize + level-0 unique ----
  k_fill_u32<<<1, 64, 0, stream>>>(mm, 0xFFFFFFFFu, 3);
  k_fill_u32<<<1, 64, 0, stream>>>(mm + 3, 0u, 3);
  k_fill_u32<<<1, 64, 0, stream>>>((unsigned*)zrow, 0u, 64);
  k_minmax<<<N / 256, 256, 0, stream>>>(points, mm, N);

  k_fill_u32<<<256, 256, 0, stream>>>((unsigned*)table, (unsigned)EMPTY_SLOT, 65536);
  k_fill_u32<<<256, 256, 0, stream>>>((unsigned*)hcnt, 0u, 65536);
  k_fill_u32<<<1, 64, 0, stream>>>((unsigned*)counter, 0u, 1);
  k_fill_u64<<<128, 256, 0, stream>>>(keys, ~0ull, N);
  k_point_ids<<<N / 256, 256, 0, stream>>>(points, batch, mm, N, G0, table, hcnt, 65536);
  k_collect<<<256, 256, 0, stream>>>(table, 65536, keys, counter, N);
  bitonic_sort(keys, N, stream);
  k_extract<<<128, 256, 0, stream>>>(keys, N, INV0, ids0, cnt0, hcnt);

  // ---- conv1 (ci=1 -> 16) + adanorm ----
  k_conv_c1<<<N / 256, 256, 0, stream>>>(ids0, N, G0, INV0, cnt0, w1, b1, h0);
  k_adanorm_relu<<<N / 8, 256, 0, stream>>>(h0, g1, N, 16);

  // ---- down1 ids ----
  k_fill_u32<<<256, 256, 0, stream>>>((unsigned*)table, (unsigned)EMPTY_SLOT, 65536);
  k_fill_u32<<<1, 64, 0, stream>>>((unsigned*)counter, 0u, 1);
  k_fill_u64<<<128, 256, 0, stream>>>(keys, ~0ull, CAP1);
  k_down_cand<<<N / 256, 256, 0, stream>>>(ids0, N, G0, G1, INV0, table, 65536);
  k_collect<<<256, 256, 0, stream>>>(table, 65536, keys, counter, CAP1);
  bitonic_sort(keys, CAP1, stream);
  k_extract<<<128, 256, 0, stream>>>(keys, CAP1, INV1, ids1, (float*)nullptr, hcnt);

  // ---- down1 gather-GEMM (16 -> 32) ----
  k_wswz<<<(27 * 1 * 2 * 32 + 255) / 256, 256, 0, stream>>>(wd1, 16, 32, 1, 2, Wb1);
  k_gather_gemm<16, 32><<<CAP1 / 128, 256, 0, stream>>>(
      ids1, CAP1, G1, ids0, N, G0, h0, zrow, Wb1, bd1, 1, INV1, h1);
  k_adanorm_relu<<<CAP1 / 8, 256, 0, stream>>>(h1, g2, CAP1, 32);

  // ---- subm2 (32 -> 32) ----
  k_wswz<<<(27 * 1 * 2 * 32 + 255) / 256, 256, 0, stream>>>(w2, 32, 32, 1, 2, Wb2);
  k_gather_gemm<32, 32><<<CAP1 / 128, 256, 0, stream>>>(
      ids1, CAP1, G1, ids1, CAP1, G1, h1, zrow, Wb2, b2, 0, INV1, h2);
  k_adanorm_relu<<<CAP1 / 8, 256, 0, stream>>>(h2, g3, CAP1, 32);

  // ---- down2 ids ----
  k_fill_u32<<<512, 256, 0, stream>>>((unsigned*)table, (unsigned)EMPTY_SLOT, 131072);
  k_fill_u32<<<1, 64, 0, stream>>>((unsigned*)counter, 0u, 1);
  k_fill_u64<<<256, 256, 0, stream>>>(keys, ~0ull, CAP2);
  k_down_cand<<<CAP1 / 256, 256, 0, stream>>>(ids1, CAP1, G1, G2, INV1, table, 131072);
  k_collect<<<512, 256, 0, stream>>>(table, 131072, keys, counter, CAP2);
  bitonic_sort(keys, CAP2, stream);
  k_extract<<<256, 256, 0, stream>>>(keys, CAP2, INV2, ids2, (float*)nullptr, hcnt);

  // ---- down2 gather-GEMM (32 -> 64) ----
  k_wswz<<<(27 * 1 * 4 * 32 + 255) / 256, 256, 0, stream>>>(wd2, 32, 64, 1, 4, Wb3);
  k_gather_gemm<32, 64><<<CAP2 / 128, 256, 0, stream>>>(
      ids2, CAP2, G2, ids1, CAP1, G1, h2, zrow, Wb3, bd2, 1, INV2, h3);
  k_adanorm_relu<<<CAP2 / 8, 256, 0, stream>>>(h3, g4, CAP2, 64);

  // ---- subm3 (64 -> 64) ----
  k_wswz<<<(27 * 2 * 4 * 32 + 255) / 256, 256, 0, stream>>>(w3, 64, 64, 2, 4, Wb4);
  k_gather_gemm<64, 64><<<CAP2 / 128, 256, 0, stream>>>(
      ids2, CAP2, G2, ids2, CAP2, G2, h3, zrow, Wb4, b3, 0, INV2, h4);
  k_adanorm_relu<<<CAP2 / 8, 256, 0, stream>>>(h4, g5, CAP2, 64);

  // ---- pool + head ----
  k_fill_u32<<<1, 256, 0, stream>>>(pooled, 0u, 256);
  k_pool<<<CAP2 / 256, 256, 0, stream>>>(ids2, CAP2, INV2, 64, h4, pooled, G2);
  k_head<<<1, 128, 0, stream>>>((const float*)pooled, wh, bh, gh, wf1, bf1, wf2, bf2,
                                (float*)d_out);
}
